// DiTBlock_54013508715100
// MI455X (gfx1250) — compile-verified, matrix operand order checked by reading, not run on hardware
//
#include <hip/hip_runtime.h>
#include <hip/hip_bf16.h>

typedef __bf16 bf16_t;
typedef __attribute__((ext_vector_type(16))) __bf16 v16bf;
typedef __attribute__((ext_vector_type(8)))  __bf16 v8bf;
typedef __attribute__((ext_vector_type(8)))  float  v8f;

#define WMMA_BF16(Afrag, Bfrag, Cfrag) \
  __builtin_amdgcn_wmma_f32_16x16x32_bf16(false, (Afrag), false, (Bfrag), (short)0, (Cfrag), false, false)

__device__ __forceinline__ v16bf load_frag(const bf16_t* rowbase, int kh) {
  // 16x32 bf16 fragment, per ISA layout: lane holds row (lane%16); halves
  // group0 at K = kh..kh+7, group1 at K = 16+kh..16+kh+7, kh = 8*(lane>=16)
  v8bf lo = *(const v8bf*)(rowbase + kh);
  v8bf hi = *(const v8bf*)(rowbase + 16 + kh);
  return __builtin_shufflevector(lo, hi, 0,1,2,3,4,5,6,7,8,9,10,11,12,13,14,15);
}

// Async global->LDS copy of 32 bytes per lane (two B128 transfers).
// INST_OFFSET applies to both the LDS and global address (ISA 08 §4.4).
__device__ __forceinline__ void async_cp32(const bf16_t* lds_dst, const bf16_t* gsrc) {
  unsigned la = (unsigned)(uintptr_t)lds_dst;           // low 32 bits = LDS offset
  unsigned long long ga = (unsigned long long)(uintptr_t)gsrc;
  asm volatile("global_load_async_to_lds_b128 %0, %1, off" :: "v"(la), "v"(ga) : "memory");
  asm volatile("global_load_async_to_lds_b128 %0, %1, off offset:16" :: "v"(la), "v"(ga) : "memory");
}

__device__ __forceinline__ float siluf(float x) { return x / (1.f + __expf(-x)); }

// ---------------------------------------------------------------------------
// adaLN modulation: mod[b, col] = silu(c[b,:]) . ada_w[col,:] + ada_b[col]
// grid (7168/256, B), block 256
// ---------------------------------------------------------------------------
__global__ __launch_bounds__(256) void k_mod(const float* __restrict__ c,
                                             const float* __restrict__ ada_w,
                                             const float* __restrict__ ada_b,
                                             float* __restrict__ mod) {
  __shared__ float cs[1024];
  int b = blockIdx.y;
  for (int i = threadIdx.x; i < 1024; i += 256) cs[i] = siluf(c[b * 1024 + i]);
  __syncthreads();
  int col = blockIdx.x * 256 + threadIdx.x;
  const float* wr = ada_w + (size_t)col * 1024;
  float acc = 0.f;
  for (int k = 0; k < 1024; ++k) acc += cs[k] * wr[k];
  mod[(size_t)b * 7168 + col] = acc + ada_b[col];
}

// ---------------------------------------------------------------------------
// h = rmsnorm(x,w) * (1 + sc) + sh  -> bf16.  mod==nullptr -> plain rmsnorm.
// grid = B*N rows, block 256 (4 elements/thread)
// ---------------------------------------------------------------------------
__global__ __launch_bounds__(256) void k_rms_mod(const float* __restrict__ x,
                                                 const float* __restrict__ w,
                                                 const float* __restrict__ mod,
                                                 int sh_off, int sc_off,
                                                 bf16_t* __restrict__ out) {
  int row = blockIdx.x;
  int b = row >> 10;
  int lane = threadIdx.x & 31, wid = threadIdx.x >> 5;
  const float* xr = x + (size_t)row * 1024;
  int base = threadIdx.x * 4;
  float v[4];
  float ss = 0.f;
#pragma unroll
  for (int i = 0; i < 4; ++i) { v[i] = xr[base + i]; ss += v[i] * v[i]; }
#pragma unroll
  for (int m = 1; m < 32; m <<= 1) ss += __shfl_xor(ss, m, 32);
  __shared__ float red[8];
  if (lane == 0) red[wid] = ss;
  __syncthreads();
  ss = 0.f;
#pragma unroll
  for (int i = 0; i < 8; ++i) ss += red[i];
  float rsn = rsqrtf(ss * (1.f / 1024.f) + 1e-6f);
#pragma unroll
  for (int i = 0; i < 4; ++i) {
    int d = base + i;
    float h = v[i] * rsn * w[d];
    if (mod) {
      float sc = mod[(size_t)b * 7168 + sc_off + d];
      float sh = mod[(size_t)b * 7168 + sh_off + d];
      h = h * (1.f + sc) + sh;
    }
    out[(size_t)row * 1024 + d] = (bf16_t)h;
  }
}

// ---------------------------------------------------------------------------
// Per-head rmsnorm (+optional RoPE) and relayout (rows,Din)f32 -> (B,H,Nseq,64)bf16
// grid = B*Nseq rows, block 256 = 16 heads x 16 threads (4 elems each)
// ---------------------------------------------------------------------------
__global__ __launch_bounds__(256) void k_headnorm(const float* __restrict__ in,
                                                  int in_stride, int col_off, int Nseq,
                                                  const float* __restrict__ w,
                                                  int do_norm, int do_rope,
                                                  const float* __restrict__ rc,
                                                  const float* __restrict__ rsn,
                                                  bf16_t* __restrict__ out) {
  int row = blockIdx.x;
  int b = row / Nseq;
  int n = row - b * Nseq;
  int hh = threadIdx.x >> 4;
  int t = threadIdx.x & 15;
  int d0 = t * 4;
  const float* src = in + (size_t)row * in_stride + col_off + hh * 64;
  float v[4];
#pragma unroll
  for (int i = 0; i < 4; ++i) v[i] = src[d0 + i];
  if (do_norm) {
    float ss = 0.f;
#pragma unroll
    for (int i = 0; i < 4; ++i) ss += v[i] * v[i];
#pragma unroll
    for (int m = 1; m < 16; m <<= 1) ss += __shfl_xor(ss, m, 32);
    float r = rsqrtf(ss * (1.f / 64.f) + 1e-6f);
#pragma unroll
    for (int i = 0; i < 4; ++i) v[i] = v[i] * r * w[d0 + i];
  }
  if (do_rope) {
    float nv[4];
#pragma unroll
    for (int i = 0; i < 4; ++i) {
      int d = d0 + i;
      float p = __shfl_xor(v[i], 8, 32);  // partner at d +/- 32
      float rot = (d < 32) ? -p : p;
      nv[i] = v[i] * rc[n * 64 + d] + rot * rsn[n * 64 + d];
    }
#pragma unroll
    for (int i = 0; i < 4; ++i) v[i] = nv[i];
  }
  bf16_t* dst = out + ((((size_t)b * 16 + hh) * Nseq + n) * 64 + d0);
#pragma unroll
  for (int i = 0; i < 4; ++i) dst[i] = (bf16_t)v[i];
}

// ---------------------------------------------------------------------------
// Generic GEMM: C(M,Nout) = A_bf(M,K) @ W_bf(Nout,K)^T with fused epilogue:
//   val = acc (+ bias[col]); val *= gate[(row>>10)*7168 + col]; val += resid
// 256 threads (8 waves), 128x128 block tile, 32x64 per-wave tile.
// Double-buffered LDS tiles filled by async global->LDS DMA (ASYNCcnt).
// ---------------------------------------------------------------------------
__global__ __launch_bounds__(256) void k_gemm(const bf16_t* __restrict__ A,
                                              const bf16_t* __restrict__ W,
                                              int Nout, int K,
                                              float* __restrict__ outF,
                                              bf16_t* __restrict__ outB,
                                              const float* __restrict__ bias,
                                              const float* __restrict__ gate,
                                              const float* __restrict__ resid) {
  __shared__ bf16_t As[2][128][32];
  __shared__ bf16_t Bs[2][128][32];
  int tid = threadIdx.x, lane = tid & 31, wid = tid >> 5;
  int wm = wid & 3, wn = wid >> 2;
  int bm = blockIdx.y * 128, bn = blockIdx.x * 128;
  int kh = 8 * (lane >> 4), rsel = lane & 15;
  int lr = tid >> 1, lc = (tid & 1) * 16;

  v8f acc[2][4];
#pragma unroll
  for (int i = 0; i < 2; ++i)
#pragma unroll
    for (int j = 0; j < 4; ++j)
#pragma unroll
      for (int r = 0; r < 8; ++r) acc[i][j][r] = 0.f;

  const bf16_t* agBase = A + (size_t)(bm + lr) * K + lc;
  const bf16_t* wgBase = W + (size_t)(bn + lr) * K + lc;

  // prologue: stage 0 -> buffer 0 (4 async instructions per wave)
  async_cp32(&As[0][lr][lc], agBase);
  async_cp32(&Bs[0][lr][lc], wgBase);

  int nk = K >> 5;
  for (int it = 0; it < nk; ++it) {
    int cur = it & 1;
    if (it + 1 < nk) {
      // prefetch next stage into the other buffer (its readers finished at
      // the trailing barrier of the previous iteration)
      async_cp32(&As[cur ^ 1][lr][lc], agBase + (it + 1) * 32);
      async_cp32(&Bs[cur ^ 1][lr][lc], wgBase + (it + 1) * 32);
      asm volatile("s_wait_asynccnt 0x4" ::: "memory");  // stage `it` done
    } else {
      asm volatile("s_wait_asynccnt 0x0" ::: "memory");
    }
    __syncthreads();  // stage `it` visible from all waves

    v16bf af[2], bfr[4];
#pragma unroll
    for (int i = 0; i < 2; ++i) af[i] = load_frag(&As[cur][wm * 32 + i * 16 + rsel][0], kh);
#pragma unroll
    for (int j = 0; j < 4; ++j) bfr[j] = load_frag(&Bs[cur][wn * 64 + j * 16 + rsel][0], kh);
#pragma unroll
    for (int i = 0; i < 2; ++i)
#pragma unroll
      for (int j = 0; j < 4; ++j) acc[i][j] = WMMA_BF16(af[i], bfr[j], acc[i][j]);
    __syncthreads();  // all reads of buffer `cur` done before it is re-targeted
  }

#pragma unroll
  for (int i = 0; i < 2; ++i)
#pragma unroll
    for (int j = 0; j < 4; ++j)
#pragma unroll
      for (int r = 0; r < 8; ++r) {
        int row = bm + wm * 32 + i * 16 + r + kh;
        int col = bn + wn * 64 + j * 16 + rsel;
        float v = acc[i][j][r];
        if (bias) v += bias[col];
        if (gate) v *= gate[(size_t)(row >> 10) * 7168 + col];
        if (resid) v += resid[(size_t)row * Nout + col];
        if (outF) outF[(size_t)row * Nout + col] = v;
        else outB[(size_t)row * Nout + col] = (bf16_t)v;
      }
}

// ---------------------------------------------------------------------------
// Flash attention: out(B,Nq,H*64)bf16 = softmax(Q K^T * scale [+mask]) V
// q,k,v: (B,H,Nseq,64) bf16.  128 threads: 4 waves x 16 query rows.
// grid (Nq/64, B*H).  K tile staged with async global->LDS DMA.
// ---------------------------------------------------------------------------
__global__ __launch_bounds__(128) void k_attn(const bf16_t* __restrict__ q,
                                              const bf16_t* __restrict__ kk,
                                              const bf16_t* __restrict__ vv,
                                              bf16_t* __restrict__ out,
                                              int Nq, int Nk,
                                              const unsigned char* __restrict__ mask,
                                              float scale) {
  __shared__ bf16_t Ks[32][64];
  __shared__ bf16_t Vt[64][32];
  __shared__ bf16_t Ps[4][16][32];
  int bh = blockIdx.y;
  int b = bh >> 4, h = bh & 15;
  int tid = threadIdx.x, lane = tid & 31, wid = tid >> 5;
  int kh = 8 * (lane >> 4), rsel = lane & 15;
  int m0 = blockIdx.x * 64 + wid * 16;

  // Q fragments stay in registers: 16 rows x 64 (two K-chunks of 32)
  const bf16_t* qb = q + ((size_t)bh * Nq + m0 + rsel) * 64;
  v16bf qf[2];
#pragma unroll
  for (int c = 0; c < 2; ++c) qf[c] = load_frag(qb + c * 32, kh);

  v8f of[4];
  float mx[8], sm[8];
#pragma unroll
  for (int j = 0; j < 4; ++j)
#pragma unroll
    for (int r = 0; r < 8; ++r) of[j][r] = 0.f;
#pragma unroll
  for (int r = 0; r < 8; ++r) { mx[r] = -3.0e38f; sm[r] = 0.f; }

  int lr = tid >> 2, lc0 = (tid & 3) * 16;
  for (int kv0 = 0; kv0 < Nk; kv0 += 32) {
    __syncthreads();
    // K tile via async DMA
    async_cp32(&Ks[lr][lc0], kk + ((size_t)bh * Nk + kv0 + lr) * 64 + lc0);
    // V tile staged transposed (needs per-element relayout)
    const bf16_t* vg = vv + ((size_t)bh * Nk + kv0 + lr) * 64 + lc0;
#pragma unroll
    for (int i = 0; i < 16; ++i) Vt[lc0 + i][lr] = vg[i];
    asm volatile("s_wait_asynccnt 0x0" ::: "memory");
    __syncthreads();

    // scores: two 16x16 tiles (kv columns kv0..+15, kv0+16..+31)
    v8f s[2];
#pragma unroll
    for (int t2 = 0; t2 < 2; ++t2) {
#pragma unroll
      for (int r = 0; r < 8; ++r) s[t2][r] = 0.f;
#pragma unroll
      for (int c = 0; c < 2; ++c) {
        v16bf kf = load_frag(&Ks[t2 * 16 + rsel][c * 32], kh);
        s[t2] = WMMA_BF16(qf[c], kf, s[t2]);
      }
    }
    bool mk0 = true, mk1 = true;
    if (mask) {
      mk0 = mask[(size_t)b * Nk + kv0 + rsel] != 0;
      mk1 = mask[(size_t)b * Nk + kv0 + 16 + rsel] != 0;
    }
#pragma unroll
    for (int r = 0; r < 8; ++r) {
      float a0 = s[0][r] * scale, a1 = s[1][r] * scale;
      if (!mk0) a0 = -1.0e30f;
      if (!mk1) a1 = -1.0e30f;
      float cm = fmaxf(a0, a1);
#pragma unroll
      for (int d2 = 1; d2 < 16; d2 <<= 1) cm = fmaxf(cm, __shfl_xor(cm, d2, 32));
      float mnew = fmaxf(mx[r], cm);
      float corr = __expf(mx[r] - mnew);
      float p0 = __expf(a0 - mnew), p1 = __expf(a1 - mnew);
      float ps = p0 + p1;
#pragma unroll
      for (int d2 = 1; d2 < 16; d2 <<= 1) ps += __shfl_xor(ps, d2, 32);
      sm[r] = sm[r] * corr + ps;
      mx[r] = mnew;
#pragma unroll
      for (int j = 0; j < 4; ++j) of[j][r] *= corr;
      Ps[wid][r + kh][rsel] = (bf16_t)p0;       // relayout P to A-frag via LDS
      Ps[wid][r + kh][16 + rsel] = (bf16_t)p1;
    }
    v16bf pf = load_frag(&Ps[wid][rsel][0], kh);
#pragma unroll
    for (int j = 0; j < 4; ++j) {
      v16bf vf = load_frag(&Vt[j * 16 + rsel][0], kh);
      of[j] = WMMA_BF16(pf, vf, of[j]);
    }
  }

#pragma unroll
  for (int j = 0; j < 4; ++j)
#pragma unroll
    for (int r = 0; r < 8; ++r) {
      float val = of[j][r] / sm[r];
      int row = m0 + r + kh;
      int col = h * 64 + j * 16 + rsel;
      out[((size_t)b * Nq + row) * 1024 + col] = (bf16_t)val;
    }
}

// ---------------------------------------------------------------------------
__global__ void k_cvt(const float* __restrict__ in, bf16_t* __restrict__ out, long n) {
  for (long i = (long)blockIdx.x * 256 + threadIdx.x; i < n; i += (long)gridDim.x * 256)
    out[i] = (bf16_t)in[i];
}

__global__ void k_silu_mul(bf16_t* __restrict__ a, const bf16_t* __restrict__ b, long n) {
  for (long i = (long)blockIdx.x * 256 + threadIdx.x; i < n; i += (long)gridDim.x * 256) {
    float x = (float)a[i], y = (float)b[i];
    a[i] = (bf16_t)(siluf(x) * y);
  }
}

// ---------------------------------------------------------------------------
extern "C" void kernel_launch(void* const* d_in, const int* in_sizes, int n_in,
                              void* d_out, int out_size, void* d_ws, size_t ws_size,
                              hipStream_t stream) {
  const float* x      = (const float*)d_in[0];
  const float* c      = (const float*)d_in[1];
  const float* ctx    = (const float*)d_in[2];
  const unsigned char* cmask = (const unsigned char*)d_in[3];
  const float* rc     = (const float*)d_in[4];
  const float* rs     = (const float*)d_in[5];
  const float* ada_w  = (const float*)d_in[6];
  const float* ada_b  = (const float*)d_in[7];
  const float* n1w    = (const float*)d_in[8];
  const float* qkvw   = (const float*)d_in[9];
  const float* projw  = (const float*)d_in[10];
  const float* projb  = (const float*)d_in[11];
  const float* qnw    = (const float*)d_in[12];
  const float* knw    = (const float*)d_in[13];
  const float* ncw    = (const float*)d_in[14];
  const float* cqw    = (const float*)d_in[15];
  const float* ckw    = (const float*)d_in[16];
  const float* cvw    = (const float*)d_in[17];
  const float* cprojw = (const float*)d_in[18];
  const float* cprojb = (const float*)d_in[19];
  const float* cqnw   = (const float*)d_in[20];
  const float* cknw   = (const float*)d_in[21];
  const float* n2w    = (const float*)d_in[22];
  const float* w1     = (const float*)d_in[23];
  const float* w2     = (const float*)d_in[24];
  const float* w3     = (const float*)d_in[25];
  float* out = (float*)d_out;

  const size_t MB = 1ull << 20;
  char* ws = (char*)d_ws;
  float*  qkvF  = (float*)(ws + 0);           // 96 MB (phase 1)
  float*  qF    = (float*)(ws + 0);           // 32 MB (cross-attn q, reuse)
  bf16_t* a_bf  = (bf16_t*)(ws + 0);          // 64 MB (MLP, reuse)
  bf16_t* b_bf  = (bf16_t*)(ws + 64 * MB);    // 64 MB
  bf16_t* qr    = (bf16_t*)(ws + 128 * MB);   // 16 MB
  bf16_t* kr    = (bf16_t*)(ws + 144 * MB);   // 16 MB
  bf16_t* vbp   = (bf16_t*)(ws + 160 * MB);   // 16 MB
  bf16_t* hbuf  = (bf16_t*)(ws + 176 * MB);   // 16 MB (h / o_bf / hc / o2_bf / h2)
  float*  kcF   = (float*)(ws + 192 * MB);    // 4 MB
  float*  vcF   = (float*)(ws + 196 * MB);    // 4 MB
  float*  modp  = (float*)(ws + 200 * MB);    // 0.25 MB
  bf16_t* ctxbf = (bf16_t*)(ws + 201 * MB);   // 2 MB
  bf16_t* wslot = (bf16_t*)(ws + 204 * MB);   // 8 MB weight slot

  const float scale = 0.125f;  // 64^-0.5

  // adaLN modulation
  k_mod<<<dim3(28, 8), 256, 0, stream>>>(c, ada_w, ada_b, modp);

  // ---- self attention ----
  k_rms_mod<<<8192, 256, 0, stream>>>(x, n1w, modp, 0, 1024, hbuf);
  k_cvt<<<4096, 256, 0, stream>>>(qkvw, wslot, 3072L * 1024);
  k_gemm<<<dim3(24, 64), 256, 0, stream>>>(hbuf, wslot, 3072, 1024, qkvF, nullptr,
                                           nullptr, nullptr, nullptr);
  k_headnorm<<<8192, 256, 0, stream>>>(qkvF, 3072, 0,    1024, qnw, 1, 1, rc, rs, qr);
  k_headnorm<<<8192, 256, 0, stream>>>(qkvF, 3072, 1024, 1024, knw, 1, 1, rc, rs, kr);
  k_headnorm<<<8192, 256, 0, stream>>>(qkvF, 3072, 2048, 1024, nullptr, 0, 0, rc, rs, vbp);
  k_attn<<<dim3(16, 128), 128, 0, stream>>>(qr, kr, vbp, hbuf, 1024, 1024, nullptr, scale);
  k_cvt<<<4096, 256, 0, stream>>>(projw, wslot, 1024L * 1024);
  k_gemm<<<dim3(8, 64), 256, 0, stream>>>(hbuf, wslot, 1024, 1024, out, nullptr,
                                          projb, modp + 2048 /*g_msa*/, x);

  // ---- cross attention ----
  k_rms_mod<<<8192, 256, 0, stream>>>(out, ncw, nullptr, 0, 0, hbuf);
  k_cvt<<<4096, 256, 0, stream>>>(cqw, wslot, 1024L * 1024);
  k_gemm<<<dim3(8, 64), 256, 0, stream>>>(hbuf, wslot, 1024, 1024, qF, nullptr,
                                          nullptr, nullptr, nullptr);
  k_headnorm<<<8192, 256, 0, stream>>>(qF, 1024, 0, 1024, cqnw, 1, 0, rc, rs, qr);
  k_cvt<<<4096, 256, 0, stream>>>(ctx, ctxbf, 1024L * 1024);
  k_cvt<<<4096, 256, 0, stream>>>(ckw, wslot, 1024L * 1024);
  k_gemm<<<dim3(8, 8), 256, 0, stream>>>(ctxbf, wslot, 1024, 1024, kcF, nullptr,
                                         nullptr, nullptr, nullptr);
  k_headnorm<<<1024, 256, 0, stream>>>(kcF, 1024, 0, 128, cknw, 1, 0, rc, rs, kr);
  k_cvt<<<4096, 256, 0, stream>>>(cvw, wslot, 1024L * 1024);
  k_gemm<<<dim3(8, 8), 256, 0, stream>>>(ctxbf, wslot, 1024, 1024, vcF, nullptr,
                                         nullptr, nullptr, nullptr);
  k_headnorm<<<1024, 256, 0, stream>>>(vcF, 1024, 0, 128, nullptr, 0, 0, rc, rs, vbp);
  k_attn<<<dim3(16, 128), 128, 0, stream>>>(qr, kr, vbp, hbuf, 1024, 128, cmask, scale);
  k_cvt<<<4096, 256, 0, stream>>>(cprojw, wslot, 1024L * 1024);
  k_gemm<<<dim3(8, 64), 256, 0, stream>>>(hbuf, wslot, 1024, 1024, out, nullptr,
                                          cprojb, modp + 3072 /*g_xattn*/, out);

  // ---- SwiGLU MLP ----
  k_rms_mod<<<8192, 256, 0, stream>>>(out, n2w, modp, 4096, 5120, hbuf);
  k_cvt<<<4096, 256, 0, stream>>>(w1, wslot, 4096L * 1024);
  k_gemm<<<dim3(32, 64), 256, 0, stream>>>(hbuf, wslot, 4096, 1024, nullptr, a_bf,
                                           nullptr, nullptr, nullptr);
  k_cvt<<<4096, 256, 0, stream>>>(w3, wslot, 4096L * 1024);
  k_gemm<<<dim3(32, 64), 256, 0, stream>>>(hbuf, wslot, 4096, 1024, nullptr, b_bf,
                                           nullptr, nullptr, nullptr);
  k_silu_mul<<<8192, 256, 0, stream>>>(a_bf, b_bf, 8192L * 4096);
  k_cvt<<<4096, 256, 0, stream>>>(w2, wslot, 4096L * 1024);
  k_gemm<<<dim3(8, 64), 256, 0, stream>>>(a_bf, wslot, 1024, 4096, out, nullptr,
                                          nullptr, modp + 6144 /*g_mlp*/, out);
}
